// Head_30142080483477
// MI455X (gfx1250) — matrix operand order repaired and checked by reading, not compiled, not run on hardware
//
#include <hip/hip_runtime.h>

// ---------------------------------------------------------------------------
// Fused attention head for MI455X (gfx1250, wave32, WMMA).
//   q = x@Wq; k = x@Wk; v = x@Wv            (bf16 WMMA, f32 accumulate)
//   logits = q@k^T  (NO 1/sqrt(d) scaling)  (bf16 WMMA)
//   masked online softmax (mask over keys, -1e9 fill)
//   out = softmax(logits)@v                 (bf16 WMMA)
//
// WMMA fragment layouts (wave32, 16x16x32 bf16):
//   A: row = lane&15; slot  e   -> K=(lane>>4)*16+e   (contiguous, same order as B)
//   B: col = lane&15; slot  e   -> K=(lane>>4)*16+e   (contiguous 32B chunk; this code's order, not AMD's table order)
//   C/D: col = lane&15; vgpr v  -> row = v + (lane>>4)*8
// ---------------------------------------------------------------------------

typedef __attribute__((ext_vector_type(16))) __bf16 v16bf;
typedef __attribute__((ext_vector_type(8)))  float  v8f;
typedef int v4i_vs __attribute__((vector_size(16)));  // matches builtin param type

union V16 {
  v16bf v;
  uint4 q[2];
  unsigned short u[16];
};

__device__ __forceinline__ unsigned short f32_bf16(float f) {
  union { __bf16 h; unsigned short u; } cv;
  cv.h = (__bf16)f;  // native v_cvt, RNE
  return cv.u;
}

// ---- gfx1250 async global->LDS copy (ASYNCcnt) with portable fallback -----
#if __has_builtin(__builtin_amdgcn_global_load_async_to_lds_b128)
#define ATHENA_ASYNC 1
#else
#define ATHENA_ASYNC 0
#endif

__device__ __forceinline__ void cp_async_b128(const void* g, void* l) {
#if ATHENA_ASYNC
  __builtin_amdgcn_global_load_async_to_lds_b128(
      (__attribute__((address_space(1))) v4i_vs*)(unsigned long long)g,
      (__attribute__((address_space(3))) v4i_vs*)(unsigned int)(unsigned long long)l,
      0, 0);
#else
  *(uint4*)l = *(const uint4*)g;
#endif
}

__device__ __forceinline__ void cp_async_wait() {
#if ATHENA_ASYNC
#if __has_builtin(__builtin_amdgcn_s_wait_asynccnt)
  __builtin_amdgcn_s_wait_asynccnt(0);
#else
  asm volatile("s_wait_asynccnt 0x0" ::: "memory");
#endif
#endif
}

#define D_E     1024
#define D_Q     512
#define TT      2048
#define NB      8
#define BT_ROWS (NB * TT)   // 16384
#define BQ      32          // query rows per block
#define BK      32          // keys per tile

// ---------------------------------------------------------------------------
// Kernel 1: C(16384x512) = bf16( X(16384x1024,f32) * W(1024x512,f32) )
// 128x128 tile per block, 8 waves, 8 accumulator tiles per wave.
// ---------------------------------------------------------------------------
__global__ __launch_bounds__(256) void proj_kernel(
    const float* __restrict__ x, const float* __restrict__ Wq,
    const float* __restrict__ Wk, const float* __restrict__ Wv,
    unsigned short* __restrict__ qkv) {
  __shared__ unsigned short Xs[128 * 32];  // [row][k]  bf16
  __shared__ unsigned short Wt[128 * 32];  // [col][k]  bf16 (transposed W tile)

  const float* W = (blockIdx.z == 0) ? Wq : (blockIdx.z == 1) ? Wk : Wv;
  unsigned short* out = qkv + (size_t)blockIdx.z * BT_ROWS * D_Q;

  const int m0 = blockIdx.x * 128;
  const int n0 = blockIdx.y * 128;
  const int tid  = threadIdx.x;
  const int wave = tid >> 5, lane = tid & 31;
  const int h = lane >> 4, nn = lane & 15;

  v8f acc[8];
#pragma unroll
  for (int j = 0; j < 8; ++j) acc[j] = {};

  for (int kb = 0; kb < D_E; kb += 32) {
#pragma unroll
    for (int i = 0; i < 16; ++i) {
      int idx = tid + i * 256;
      int r = idx >> 5, c = idx & 31;
      Xs[r * 32 + c] = f32_bf16(x[(size_t)(m0 + r) * D_E + kb + c]);
      int k = idx >> 7, nc = idx & 127;
      Wt[nc * 32 + k] = f32_bf16(W[(size_t)(kb + k) * D_Q + n0 + nc]);
    }
    __syncthreads();

    V16 a;  // A fragment: row = wave*16 + nn, K = h*16 + e (contiguous)
    a.q[0] = *(const uint4*)&Xs[(wave * 16 + nn) * 32 + h * 16];
    a.q[1] = *(const uint4*)&Xs[(wave * 16 + nn) * 32 + h * 16 + 8];
#pragma unroll
    for (int ct = 0; ct < 8; ++ct) {
      V16 bm;  // B fragment: col = ct*16 + nn, K = h*16 + e (contiguous)
      bm.q[0] = *(const uint4*)&Wt[(ct * 16 + nn) * 32 + h * 16];
      bm.q[1] = *(const uint4*)&Wt[(ct * 16 + nn) * 32 + h * 16 + 8];
      acc[ct] = __builtin_amdgcn_wmma_f32_16x16x32_bf16(
          false, a.v, false, bm.v, (short)0, acc[ct], false, false);
    }
    __syncthreads();
  }

#pragma unroll
  for (int ct = 0; ct < 8; ++ct)
#pragma unroll
    for (int v = 0; v < 8; ++v) {
      int row = m0 + wave * 16 + v + h * 8;
      int col = n0 + ct * 16 + nn;
      out[(size_t)row * D_Q + col] = f32_bf16(acc[ct][v]);
    }
}

// ---------------------------------------------------------------------------
// Kernel 2: flash attention over one batch / 32-query tile per block.
// ---------------------------------------------------------------------------
__global__ __launch_bounds__(256) void attn_kernel(
    const unsigned short* __restrict__ qbf, const unsigned short* __restrict__ kbf,
    const unsigned short* __restrict__ vbf, const unsigned char* __restrict__ mask,
    float* __restrict__ out) {
  extern __shared__ char smem[];
  unsigned short* Qs = (unsigned short*)smem;          // 32*512 bf16 [q][d]
  unsigned short* Ks = Qs + BQ * D_Q;                  // 32*512 bf16 [key][d]
  unsigned short* Vt = Ks + BK * D_Q;                  // 512*32 bf16 [col][s] (transposed!)
  float*          Ss = (float*)(Vt + D_Q * BK);        // 32*32 f32 scores
  unsigned short* Ps = (unsigned short*)(Ss + BQ * BK);// 32*32 bf16 probs
  float*     scale_s = (float*)(Ps + BQ * BK);         // 32 running rescale
  float*         l_s = scale_s + BQ;                   // 32 running denom
  unsigned char*  ms = (unsigned char*)(l_s + BQ);     // 32 mask bytes

  const int b  = blockIdx.y;
  const int q0 = blockIdx.x * BQ;
  const int tid  = threadIdx.x;
  const int wave = tid >> 5, lane = tid & 31;
  const int h = lane >> 4, nn = lane & 15;
  const int rt = wave & 1;               // O row tile (0..1)
  const int cbase = (wave >> 1) * 128;   // O col base (8 col tiles per wave)

  // Stage Q tile once (async direct-to-LDS).
  const unsigned short* Qg = qbf + ((size_t)b * TT + q0) * D_Q;
#pragma unroll
  for (int i = 0; i < 8; ++i) {
    int idx = tid + i * 256;  // uint4 chunk id
    cp_async_b128(Qg + idx * 8, Qs + idx * 8);
  }

  float m_run = -1.0e30f, l_run = 0.f;
  v8f oacc[8];
#pragma unroll
  for (int j = 0; j < 8; ++j) oacc[j] = {};

  for (int kt = 0; kt < TT / BK; ++kt) {
    const int key0 = kt * BK;
    const unsigned short* Kg = kbf + ((size_t)b * TT + key0) * D_Q;
    const unsigned short* Vg = vbf + ((size_t)b * TT + key0) * D_Q;

    // K: async global->LDS, row-major.
#pragma unroll
    for (int i = 0; i < 8; ++i) {
      int idx = tid + i * 256;
      cp_async_b128(Kg + idx * 8, Ks + idx * 8);
    }
    // V: through VGPRs, transposed into Vt[col][s] so the P*V B-operand is
    // a contiguous 32B chunk per lane.
#pragma unroll
    for (int i = 0; i < 8; ++i) {
      int idx = tid + i * 256;      // chunk: row s = idx>>6, cols (idx&63)*8..+7
      int s = idx >> 6;
      int c0 = (idx & 63) * 8;
      uint4 d = *(const uint4*)(Vg + idx * 8);
      const unsigned short* u = (const unsigned short*)&d;
#pragma unroll
      for (int j = 0; j < 8; ++j) Vt[(c0 + j) * BK + s] = u[j];
    }
    if (tid < BK) ms[tid] = mask[(size_t)b * TT + key0 + tid];
    if (tid == 0 && kt + 1 < TT / BK)  // global_prefetch_b8 on next K tile
      __builtin_prefetch(kbf + ((size_t)b * TT + key0 + BK) * D_Q, 0, 1);
    cp_async_wait();
    __syncthreads();

    // Phase A: S(32x32) = Q * K^T   (waves 0..3, one 16x16 tile each)
    if (wave < 4) {
      const int srt = wave >> 1, sct = wave & 1;
      v8f s = {};
#pragma unroll
      for (int k0 = 0; k0 < D_Q; k0 += 32) {
        V16 a, bm;
        a.q[0]  = *(const uint4*)&Qs[(srt * 16 + nn) * D_Q + k0 + h * 16];
        a.q[1]  = *(const uint4*)&Qs[(srt * 16 + nn) * D_Q + k0 + h * 16 + 8];
        // B col n = key (sct*16+nn); K = k0 + h*16 + e -> contiguous row chunk
        bm.q[0] = *(const uint4*)&Ks[(sct * 16 + nn) * D_Q + k0 + h * 16];
        bm.q[1] = *(const uint4*)&Ks[(sct * 16 + nn) * D_Q + k0 + h * 16 + 8];
        s = __builtin_amdgcn_wmma_f32_16x16x32_bf16(
            false, a.v, false, bm.v, (short)0, s, false, false);
      }
#pragma unroll
      for (int v = 0; v < 8; ++v)
        Ss[(srt * 16 + v + h * 8) * BK + sct * 16 + nn] = s[v];
    }
    __syncthreads();

    // Phase B: masked online softmax, one row per thread (tid 0..31)
    if (tid < BQ) {
      float rmax = m_run;
      float vals[BK];
#pragma unroll
      for (int c = 0; c < BK; ++c) {
        float sv = ms[c] ? Ss[tid * BK + c] : -1.0e9f;
        vals[c] = sv;
        rmax = fmaxf(rmax, sv);
      }
      float sc = __expf(m_run - rmax);
      float lsum = 0.f;
#pragma unroll
      for (int c = 0; c < BK; ++c) {
        float p = __expf(vals[c] - rmax);
        lsum += p;
        Ps[tid * BK + c] = f32_bf16(p);
      }
      l_run = l_run * sc + lsum;
      m_run = rmax;
      scale_s[tid] = sc;
    }
    __syncthreads();

    // Phase C: O += P * V   (all 8 waves; K=32 -> single WMMA per tile)
    float rs[8];
#pragma unroll
    for (int v = 0; v < 8; ++v) rs[v] = scale_s[rt * 16 + v + h * 8];
    V16 a;
    a.q[0] = *(const uint4*)&Ps[(rt * 16 + nn) * BK + h * 16];
    a.q[1] = *(const uint4*)&Ps[(rt * 16 + nn) * BK + h * 16 + 8];
#pragma unroll
    for (int j = 0; j < 8; ++j) {
#pragma unroll
      for (int v = 0; v < 8; ++v) oacc[j][v] *= rs[v];
      V16 bm;  // B col n = output col; K = s = h*16 + e -> contiguous in Vt
      const int col = cbase + j * 16 + nn;
      bm.q[0] = *(const uint4*)&Vt[col * BK + h * 16];
      bm.q[1] = *(const uint4*)&Vt[col * BK + h * 16 + 8];
      oacc[j] = __builtin_amdgcn_wmma_f32_16x16x32_bf16(
          false, a.v, false, bm.v, (short)0, oacc[j], false, false);
    }
    __syncthreads();
  }

  if (tid < BQ) l_s[tid] = l_run;
  __syncthreads();
#pragma unroll
  for (int v = 0; v < 8; ++v) {
    const int rloc = rt * 16 + v + h * 8;
    const float rinv = 1.0f / l_s[rloc];
#pragma unroll
    for (int j = 0; j < 8; ++j) {
      const int col = cbase + j * 16 + nn;
      out[((size_t)b * TT + q0 + rloc) * D_Q + col] = oacc[j][v] * rinv;
    }
  }
}

// ---------------------------------------------------------------------------
extern "C" void kernel_launch(void* const* d_in, const int* in_sizes, int n_in,
                              void* d_out, int out_size, void* d_ws, size_t ws_size,
                              hipStream_t stream) {
  const float* x  = (const float*)d_in[0];
  const float* Wq = (const float*)d_in[1];
  const float* Wk = (const float*)d_in[2];
  const float* Wv = (const float*)d_in[3];
  const unsigned char* mask = (const unsigned char*)d_in[4];
  float* out = (float*)d_out;

  unsigned short* qkv = (unsigned short*)d_ws;  // 3 planes of 16384x512 bf16
  const size_t plane = (size_t)BT_ROWS * D_Q;

  dim3 g1(BT_ROWS / 128, D_Q / 128, 3);  // (128, 4, 3)
  proj_kernel<<<g1, 256, 0, stream>>>(x, Wq, Wk, Wv, qkv);

  const size_t shm = (size_t)(3 * BQ * D_Q) * 2   // Q,K tiles + Vt (bf16)
                   + (size_t)(BQ * BK) * 4        // S (f32)
                   + (size_t)(BQ * BK) * 2        // P (bf16)
                   + (size_t)BQ * 4 * 2           // scale + l
                   + (size_t)BK;                  // mask bytes
  (void)hipFuncSetAttribute((const void*)attn_kernel,
                            hipFuncAttributeMaxDynamicSharedMemorySize, (int)shm);
  dim3 g2(TT / BQ, NB);  // (64, 8)
  attn_kernel<<<g2, 256, shm, stream>>>(qkv, qkv + plane, qkv + 2 * plane, mask, out);
}
